// Multimodal_Attention_70635032150372
// MI455X (gfx1250) — matrix operand order repaired and checked by reading, not compiled, not run on hardware
//
#include <hip/hip_runtime.h>
#include <math.h>

typedef __attribute__((ext_vector_type(16))) __bf16 v16bf;
typedef __attribute__((ext_vector_type(8)))  __bf16 v8bf;
typedef __attribute__((ext_vector_type(4)))  __bf16 v4bf;
typedef __attribute__((ext_vector_type(8)))  float  v8f;

#define N_TOTAL   131072
#define C_DIM     256
#define ROWS_BLK  64          // rows of the batch handled per block (4 x 16-row M-tiles)
#define SROW      264         // padded LDS row stride in bf16 elements (bank decorrelation)
#define THREADS   256         // 8 wave32 waves: wave w -> M-tile (w&3), j-half (w>>2)
#define WELEMS    (256 * 256) // elements per weight matrix

// ---------------------------------------------------------------------------
// Prologue: split the three f32 weight matrices into bf16 hi/lo pairs in d_ws.
// ws layout (bf16 elements): [0]=W_sh_hi [1]=W_sh_lo [2]=W_img_hi [3]=W_img_lo
//                            [4]=W_ph_hi [5]=W_ph_lo ; then 3 f32 reduction slots.
// ---------------------------------------------------------------------------
__global__ void split_weights_kernel(const float* __restrict__ Wsh,
                                     const float* __restrict__ Wim,
                                     const float* __restrict__ Wph,
                                     __bf16* __restrict__ ws,
                                     float* __restrict__ sums)
{
    int i = blockIdx.x * blockDim.x + threadIdx.x;   // [0, 3*65536)
    if (i < 3 * WELEMS) {
        int mtx = i >> 16;
        int e   = i & (WELEMS - 1);
        const float* W = (mtx == 0) ? Wsh : (mtx == 1) ? Wim : Wph;
        float w = W[e];
        __bf16 hi = (__bf16)w;
        __bf16 lo = (__bf16)(w - (float)hi);
        ws[(size_t)(2 * mtx + 0) * WELEMS + e] = hi;
        ws[(size_t)(2 * mtx + 1) * WELEMS + e] = lo;
    }
    if (i < 3) sums[i] = 0.0f;
}

// ---------------------------------------------------------------------------
// WMMA helper: D = A(16x32 bf16) * B(32x16 bf16) + C(f32)
// ---------------------------------------------------------------------------
__device__ inline v8f wmma_bf16(v16bf a, v16bf b, v8f c) {
    return __builtin_amdgcn_wmma_f32_16x16x32_bf16(
        /*neg_a=*/false, a, /*neg_b=*/false, b,
        /*c_mod=*/(short)0, c, /*reuse_a=*/false, /*reuse_b=*/false);
}

// A-matrix (16x32 bf16) per-lane fetch from an LDS row.
// Layout: lane half h: elements 0..15 = K = k0+16h .. +15 (contiguous, matches B)
__device__ inline v16bf ldA(const __bf16* rowp, int koff) {
    v8bf lo = *(const v8bf*)(rowp + koff);
    v8bf hi = *(const v8bf*)(rowp + koff + 8);
    v16bf r;
#pragma unroll
    for (int i = 0; i < 8; ++i) { r[i] = lo[i]; r[8 + i] = hi[i]; }
    return r;
}

// ---------------------------------------------------------------------------
// Main fused kernel: 3x GEMM (3-term bf16 emulation of f32) + bias + tanh +
// joint_embed combine + sum-of-squares reduction.
// Each block: 64 rows. Wave w: M-tile = (w&3), N-tiles j = (w>>2)*8 .. +7.
// ---------------------------------------------------------------------------
__global__ __launch_bounds__(THREADS) void fused_attention_kernel(
    const float* __restrict__ img, const float* __restrict__ ph,
    const __bf16* __restrict__ ws,
    const float* __restrict__ b_sh, const float* __restrict__ b_im,
    const float* __restrict__ b_ph,
    float* __restrict__ out, float* __restrict__ sums)
{
    extern __shared__ char smem[];
    __bf16* sA   = (__bf16*)smem;  // [6][ROWS_BLK][SROW] bf16 splits of sh/img/ph
    float*  sRed = (float*)(smem + (size_t)6 * ROWS_BLK * SROW * sizeof(__bf16));

    const int tid = threadIdx.x;
    if (tid < 3) sRed[tid] = 0.0f;

    const int blockRow = blockIdx.x * ROWS_BLK;

    // ---- Stage activations into LDS as bf16 hi/lo splits (row-major) ----
#pragma unroll
    for (int it = 0; it < (ROWS_BLK * C_DIM / 4) / THREADS; ++it) {
        int e4  = it * THREADS + tid;   // 4 columns per step
        int row = e4 >> 6;              // 64 quads per row
        int c4  = (e4 & 63) * 4;
        const float4 xi = *(const float4*)(img + (size_t)(blockRow + row) * C_DIM + c4);
        const float4 xp = *(const float4*)(ph  + (size_t)(blockRow + row) * C_DIM + c4);
        float vi[4] = {xi.x, xi.y, xi.z, xi.w};
        float vp[4] = {xp.x, xp.y, xp.z, xp.w};
        v4bf shH, shL, imH, imL, phH, phL;
#pragma unroll
        for (int q = 0; q < 4; ++q) {
            float s = 0.5f * (vi[q] + vp[q]);
            __bf16 h;
            h = (__bf16)s;      shH[q] = h; shL[q] = (__bf16)(s     - (float)h);
            h = (__bf16)vi[q];  imH[q] = h; imL[q] = (__bf16)(vi[q] - (float)h);
            h = (__bf16)vp[q];  phH[q] = h; phL[q] = (__bf16)(vp[q] - (float)h);
        }
        size_t base = (size_t)row * SROW + c4;
        *(v4bf*)(sA + (size_t)0 * ROWS_BLK * SROW + base) = shH;
        *(v4bf*)(sA + (size_t)1 * ROWS_BLK * SROW + base) = shL;
        *(v4bf*)(sA + (size_t)2 * ROWS_BLK * SROW + base) = imH;
        *(v4bf*)(sA + (size_t)3 * ROWS_BLK * SROW + base) = imL;
        *(v4bf*)(sA + (size_t)4 * ROWS_BLK * SROW + base) = phH;
        *(v4bf*)(sA + (size_t)5 * ROWS_BLK * SROW + base) = phL;
    }
    __syncthreads();

    const int lane     = tid & 31;
    const int wv       = tid >> 5;
    const int m        = lane & 15;   // A row within tile / B-N column / C-N column
    const int h        = lane >> 4;   // lane half
    const int rtile    = wv & 3;      // 4 M-tiles of 16 rows
    const int jbase    = (wv >> 2) * 8;
    const int rowoff   = rtile * 16 + m;
    const int rowstart = blockRow + rtile * 16;

    const __bf16* aShH = sA + (size_t)(0 * ROWS_BLK + rowoff) * SROW;
    const __bf16* aShL = sA + (size_t)(1 * ROWS_BLK + rowoff) * SROW;
    const __bf16* aImH = sA + (size_t)(2 * ROWS_BLK + rowoff) * SROW;
    const __bf16* aImL = sA + (size_t)(3 * ROWS_BLK + rowoff) * SROW;
    const __bf16* aPhH = sA + (size_t)(4 * ROWS_BLK + rowoff) * SROW;
    const __bf16* aPhL = sA + (size_t)(5 * ROWS_BLK + rowoff) * SROW;

    float ssS = 0.f, ssI = 0.f, ssP = 0.f;

    for (int jt = 0; jt < 8; ++jt) {
        const int j   = jbase + jt;
        const int col = j * 16 + m;     // output column; also W row (x @ W.T)
        // B-layout: lane = column n, lane-half selects K half; K contiguous in W row
        const __bf16* bShH = ws + (size_t)0 * WELEMS + (size_t)col * C_DIM;
        const __bf16* bShL = ws + (size_t)1 * WELEMS + (size_t)col * C_DIM;
        const __bf16* bImH = ws + (size_t)2 * WELEMS + (size_t)col * C_DIM;
        const __bf16* bImL = ws + (size_t)3 * WELEMS + (size_t)col * C_DIM;
        const __bf16* bPhH = ws + (size_t)4 * WELEMS + (size_t)col * C_DIM;
        const __bf16* bPhL = ws + (size_t)5 * WELEMS + (size_t)col * C_DIM;

        v8f accS = {}, accI = {}, accP = {};

#pragma unroll 2
        for (int kk = 0; kk < 8; ++kk) {
            const int k0 = kk * 32;
            const int ka = k0 + 16 * h;   // A lane offset
            const int kb = k0 + 16 * h;   // B lane offset
            v16bf xShH = ldA(aShH, ka), xShL = ldA(aShL, ka);
            v16bf xImH = ldA(aImH, ka), xImL = ldA(aImL, ka);
            v16bf xPhH = ldA(aPhH, ka), xPhL = ldA(aPhL, ka);
            v16bf wShH = *(const v16bf*)(bShH + kb);
            v16bf wShL = *(const v16bf*)(bShL + kb);
            v16bf wImH = *(const v16bf*)(bImH + kb);
            v16bf wImL = *(const v16bf*)(bImL + kb);
            v16bf wPhH = *(const v16bf*)(bPhH + kb);
            v16bf wPhL = *(const v16bf*)(bPhL + kb);
            // 3-term f32 emulation: hi*hi + hi*lo + lo*hi
            accS = wmma_bf16(xShH, wShH, accS);
            accS = wmma_bf16(xShH, wShL, accS);
            accS = wmma_bf16(xShL, wShH, accS);
            accI = wmma_bf16(xImH, wImH, accI);
            accI = wmma_bf16(xImH, wImL, accI);
            accI = wmma_bf16(xImL, wImH, accI);
            accP = wmma_bf16(xPhH, wPhH, accP);
            accP = wmma_bf16(xPhH, wPhL, accP);
            accP = wmma_bf16(xPhL, wPhH, accP);
        }

        const float bS = b_sh[col], bI = b_im[col], bP = b_ph[col];
        // C/D layout: VGPR v, lanes 0-15 -> M=v, lanes 16-31 -> M=v+8
#pragma unroll
        for (int v = 0; v < 8; ++v) {
            int    row = rowstart + v + 8 * h;
            size_t idx = (size_t)row * C_DIM + col;
            float xi = img[idx], xp = ph[idx];   // L2 hit: staged by this block
            float xs = 0.5f * (xi + xp);
            float tS = tanhf(accS[v] + bS);
            float tI = tanhf(accI[v] + bI);
            float tP = tanhf(accP[v] + bP);
            ssS += tS * tS; ssI += tI * tI; ssP += tP * tP;
            out[idx] = tS * xs + tI * xi + tP * xp;
        }
    }

    // ---- wave32 reduction, then block, then global atomics ----
#pragma unroll
    for (int off = 16; off > 0; off >>= 1) {
        ssS += __shfl_xor(ssS, off, 32);
        ssI += __shfl_xor(ssI, off, 32);
        ssP += __shfl_xor(ssP, off, 32);
    }
    if (lane == 0) {
        atomicAdd(&sRed[0], ssS);
        atomicAdd(&sRed[1], ssI);
        atomicAdd(&sRed[2], ssP);
    }
    __syncthreads();
    if (tid == 0) {
        atomicAdd(&sums[0], sRed[0]);
        atomicAdd(&sums[1], sRed[1]);
        atomicAdd(&sums[2], sRed[2]);
    }
}

// ---------------------------------------------------------------------------
// Final scalar kernel: softmax over [img_score, ph_score]
// ---------------------------------------------------------------------------
__global__ void scores_kernel(const float* __restrict__ sums, float* __restrict__ out)
{
    float denom = sums[0] + 1e-8f;
    float is = sums[1] / denom;
    float ps = sums[2] / denom;
    float mx = fmaxf(is, ps);
    float ei = expf(is - mx), ep = expf(ps - mx);
    float inv = 1.0f / (ei + ep);
    out[(size_t)N_TOTAL * C_DIM + 0] = ei * inv;
    out[(size_t)N_TOTAL * C_DIM + 1] = ep * inv;
}

extern "C" void kernel_launch(void* const* d_in, const int* in_sizes, int n_in,
                              void* d_out, int out_size, void* d_ws, size_t ws_size,
                              hipStream_t stream) {
    const float* img = (const float*)d_in[0];
    const float* ph  = (const float*)d_in[1];
    const float* Wsh = (const float*)d_in[2];
    const float* bsh = (const float*)d_in[3];
    const float* Wim = (const float*)d_in[4];
    const float* bim = (const float*)d_in[5];
    const float* Wph = (const float*)d_in[6];
    const float* bph = (const float*)d_in[7];
    float*  out    = (float*)d_out;
    __bf16* wsplit = (__bf16*)d_ws;
    float*  sums   = (float*)((char*)d_ws + (size_t)6 * WELEMS * sizeof(__bf16));

    // 1) weight bf16 hi/lo split + zero reduction slots
    split_weights_kernel<<<(3 * WELEMS) / 256, 256, 0, stream>>>(Wsh, Wim, Wph, wsplit, sums);

    // 2) fused GEMM/tanh/combine (dynamic LDS: 6 split tiles + 3 reduction floats)
    const int smemBytes = 6 * ROWS_BLK * SROW * (int)sizeof(__bf16) + 16;
    (void)hipFuncSetAttribute((const void*)fused_attention_kernel,
                              hipFuncAttributeMaxDynamicSharedMemorySize, smemBytes);
    fused_attention_kernel<<<N_TOTAL / ROWS_BLK, THREADS, smemBytes, stream>>>(
        img, ph, wsplit, bsh, bim, bph, out, sums);

    // 3) softmax scalars
    scores_kernel<<<1, 1, 0, stream>>>(sums, out);
}
